// GATom_40750649705125
// MI455X (gfx1250) — matrix operand order repaired and checked by reading, not compiled, not run on hardware
//
#include <hip/hip_runtime.h>
#include <hip/hip_bf16.h>

// ============================================================================
// GATom GNN forward for MI455X (gfx1250), wave32 + WMMA bf16.
//
// All dense GEMMs run through v_wmma_f32_16x16x32_bf16. One wave computes a
// 16x64 output strip (4 accumulators); 8 waves/block cover 128 rows. The
// block's 64-column weight panel (bf16, transposed to [N][Kp]) is staged into
// LDS once per block (cooperative b128 loads + barrier) with an 8-half row
// pad for conflict-free ds_load_b128 fragment reads; A fragments stream from
// global with software pipelining (next K-step preloaded before the WMMAs).
// The e2@We GEMM is fused with the GATv2 logit computation (gather xl/xr,
// leaky-relu, per-head dot with att) to avoid a 491MB f32 intermediate.
//
// Input pointer order assumption (setup_inputs() dict insertion order,
// params flattened in insertion order, nested block dicts in insertion order):
//  0 x[10000x92] 1 edge_index[2x80000] 2 edge_attr[80000x50]
//  3 edge_index_line[2x160000] 4 edge_attr_line[160000x40]
//  5 batch[10000] 6 batch_line[80000] 7 global_features[128x64]
//  8 pre_node_W 9 pre_node_b 10 pre_edge_W 11 pre_edge_b
// 12 pre_edge_line_W 13 pre_edge_line_b
// 14..24 blk{norm_g,norm_b,Wl,bl,Wr,br,We,att,gat_b,gate_W,gate_b} (L=4)
// 25..35 lblk{...} (L=1)
// 36 glob_W 37 pool_gate_W 38 pool_gate_b 39 post_W 40 post_b 41 out_W 42 out_b
// ============================================================================

typedef __attribute__((ext_vector_type(16))) __bf16 v16bf;
typedef __attribute__((ext_vector_type(8)))  float  v8f;
typedef unsigned short u16;

#define ACT_NONE 0
#define ACT_SILU 1
#define ACT_ELU  2

// ---------------------------------------------------------------- constants
constexpr long  cN  = 10000;   // nodes
constexpr long  cE  = 80000;   // edges (= line-graph nodes)
constexpr long  cEL = 160000;  // line-graph edges
constexpr long  cB  = 128;     // graphs in batch
constexpr long  E2MAX = cEL + cE;   // 240000 (largest edges+selfloops)

// workspace layout (bytes, everything naturally 256B aligned)
constexpr size_t O_H     = 0;
constexpr size_t O_XLINE = O_H     + (size_t)cN  * 128 * 4;
constexpr size_t O_EA    = O_XLINE + (size_t)cE  * 128 * 4;
constexpr size_t O_EAL   = O_EA    + (size_t)cE  * 128 * 4;
constexpr size_t O_HN    = O_EAL   + (size_t)cEL * 128 * 4;
constexpr size_t O_XL    = O_HN    + (size_t)cE  * 128 * 4;
constexpr size_t O_XR    = O_XL    + (size_t)cE  * 512 * 4;  // reused as GLU gemm out
constexpr size_t O_GOUT  = O_XR    + (size_t)cE  * 512 * 4;
constexpr size_t O_ACTA  = O_GOUT  + (size_t)cE  * 512 * 4;  // bf16 A operands / e2
constexpr size_t O_CAT   = O_ACTA  + (size_t)E2MAX * 128 * 2; // bf16 [M x 640]; aliases mean_e f32
constexpr size_t O_WT    = O_CAT   + (size_t)cE * 640 * 2;    // bf16 transposed weights
constexpr size_t O_LOGIT = O_WT    + (size_t)256 * 640 * 2;
constexpr size_t O_MAXB  = O_LOGIT + (size_t)E2MAX * 4 * 4;
constexpr size_t O_SUMB  = O_MAXB  + (size_t)cE * 4 * 4;
constexpr size_t O_DEG   = O_SUMB  + (size_t)cE * 4 * 4;
constexpr size_t O_SRC2  = O_DEG   + (size_t)cE * 4;
constexpr size_t O_DST2  = O_SRC2  + (size_t)E2MAX * 4;
constexpr size_t O_POOL  = O_DST2  + (size_t)E2MAX * 4;
constexpr size_t O_POOLL = O_POOL  + (size_t)cB * 128 * 4;
constexpr size_t O_GEMB  = O_POOLL + (size_t)cB * 128 * 4;
constexpr size_t O_T2    = O_GEMB  + (size_t)cB * 128 * 4;
constexpr size_t O_S0    = O_T2    + (size_t)cB * 256 * 4;
constexpr size_t O_S1    = O_S0    + (size_t)cB * 128 * 4;
constexpr size_t O_CAT3  = O_S1    + (size_t)cB * 128 * 4;
// total ~835 MB

// ---------------------------------------------------------------- device utils
__device__ inline float act_apply(float x, int act) {
  if (act == ACT_SILU) return x / (1.f + __expf(-x));
  if (act == ACT_ELU)  return x > 0.f ? x : (__expf(x) - 1.f);
  return x;
}
__device__ inline u16 f2bf(float f) {               // round-to-nearest-even
  unsigned u = __float_as_uint(f);
  u += 0x7FFFu + ((u >> 16) & 1u);
  return (u16)(u >> 16);
}
__device__ inline v16bf make_frag(uint4 a, uint4 b) {
  union { v16bf v; uint4 q[2]; } u;
  u.q[0] = a; u.q[1] = b;
  return u.v;
}
__device__ inline void atomicMaxF(float* addr, float val) {
  unsigned* ua = (unsigned*)addr;
  unsigned old = *ua;
  while (__uint_as_float(old) < val) {
    unsigned assumed = old;
    old = atomicCAS(ua, assumed, __float_as_uint(val));
    if (old == assumed) break;
  }
}

#define GS_LOOP(i, total)                                                     \
  for (size_t i = (size_t)blockIdx.x * blockDim.x + threadIdx.x; i < (total); \
       i += (size_t)gridDim.x * blockDim.x)

// ---------------------------------------------------------------- small kernels
__global__ void fill_kernel(float* __restrict__ p, float v, size_t n) {
  GS_LOOP(i, n) p[i] = v;
}
__global__ void copy_kernel(float* __restrict__ d, const float* __restrict__ s, size_t n) {
  GS_LOOP(i, n) d[i] = s[i];
}
// dst[m*ldd + col0 + k] = bf16(src[m*lds + k] + bias[k]) for k<K, 0 for k in [K,Kp)
__global__ void cvt_rows_kernel(u16* __restrict__ dst, int ldd, int col0,
                                const float* __restrict__ src, int lds,
                                const float* __restrict__ bias,
                                size_t M, int K, int Kp) {
  GS_LOOP(i, M * (size_t)Kp) {
    size_t m = i / Kp;
    int k = (int)(i % Kp);
    u16 o = 0;
    if (k < K) {
      float v = src[m * (size_t)lds + k];
      if (bias) v += bias[k];
      o = f2bf(v);
    }
    dst[m * (size_t)ldd + col0 + k] = o;
  }
}
// dst[n*Kp + k] = bf16(src[k*N + n]); zero-pad k in [K,Kp)
__global__ void transpose_cvt_kernel(u16* __restrict__ dst, const float* __restrict__ src,
                                     int K, int N, int Kp) {
  GS_LOOP(i, (size_t)N * Kp) {
    int n = (int)(i / Kp);
    int k = (int)(i % Kp);
    dst[i] = (k < K) ? f2bf(src[(size_t)k * N + n]) : (u16)0;
  }
}

// ------------------------------------------------ shared WMMA tile machinery
// Cooperatively stage the 64-column B panel [64 x Kp] (rows = output columns
// n0..n0+63 of Wt[N][Kp]) into LDS with an 8-half row pad (bank-conflict-free
// b128 fragment reads). All 256 threads participate; caller must __syncthreads.
__device__ inline void stage_B_panel(u16* ldsB, int ldsStride, const u16* __restrict__ Wt,
                                     int n0, int Kp) {
  const int nvec = (64 * Kp) >> 3;            // 8 halves per vector
  const int kv_per_row = Kp >> 3;
  for (int v = threadIdx.x; v < nvec; v += 256) {
    const int c = v / kv_per_row;
    const int kv = v - c * kv_per_row;
    *(uint4*)&ldsB[c * ldsStride + (kv << 3)] =
        *(const uint4*)(Wt + (size_t)(n0 + c) * Kp + (kv << 3));
  }
}

// Compute the 16x64 WMMA strip: acc[j] (j = N-subtile) over A row panel at
// arow (bf16, ld = Kp) against the staged LDS B panel. A loads are software
// pipelined one K-step ahead.
__device__ inline void wmma_strip(v8f acc[4], const u16* __restrict__ arow,
                                  const u16* ldsB, int ldsStride, int Kp,
                                  int half, int r) {
  uint4 a_lo = *(const uint4*)(arow + half * 8);
  uint4 a_hi = *(const uint4*)(arow + 16 + half * 8);
  for (int k0 = 0; k0 < Kp; k0 += 32) {
    uint4 n_lo = a_lo, n_hi = a_hi;
    if (k0 + 32 < Kp) {                       // preload next K-step (pipeline)
      __builtin_prefetch(arow + k0 + 64, 0, 3);
      n_lo = *(const uint4*)(arow + k0 + 32 + half * 8);
      n_hi = *(const uint4*)(arow + k0 + 48 + half * 8);
    }
    const v16bf af = make_frag(a_lo, a_hi);
#pragma unroll
    for (int j = 0; j < 4; ++j) {
      const u16* bp = &ldsB[(j * 16 + r) * ldsStride + k0 + half * 8];
      const v16bf bfr = make_frag(*(const uint4*)bp, *(const uint4*)(bp + 16));
      acc[j] = __builtin_amdgcn_wmma_f32_16x16x32_bf16(
          false, af, false, bfr, (short)0, acc[j], false, false);
    }
    a_lo = n_lo; a_hi = n_hi;
  }
}

// ---------------------------------------------------------------- WMMA GEMM
// C[M x N] = act(A[M x Kp](bf16, ld=Kp) @ Wt[N x Kp]^T(bf16) + bias[N])
// grid = (N/64, ceil(M/128)), block = 256 (8 waves); dynamic LDS = 64*(Kp+8)*2.
__global__ __launch_bounds__(256)
void wmma_gemm_kernel(const u16* __restrict__ A, const u16* __restrict__ Wt,
                      const float* __restrict__ bias, float* __restrict__ C,
                      int M, int N, int Kp, int act) {
  extern __shared__ u16 ldsB[];
  const int ldsStride = Kp + 8;
  const int n0 = blockIdx.x << 6;
  stage_B_panel(ldsB, ldsStride, Wt, n0, Kp);
  __syncthreads();

  const int wave = threadIdx.x >> 5, lane = threadIdx.x & 31;
  const int half = lane >> 4, r = lane & 15;
  const int m0 = (blockIdx.y * 8 + wave) << 4;
  if (m0 >= M) return;

  const v8f z = {0.f, 0.f, 0.f, 0.f, 0.f, 0.f, 0.f, 0.f};
  v8f acc[4] = {z, z, z, z};
  wmma_strip(acc, A + (size_t)(m0 + r) * Kp, ldsB, ldsStride, Kp, half, r);

#pragma unroll
  for (int j = 0; j < 4; ++j) {
    const int col = n0 + j * 16 + r;
    const float bv = bias ? bias[col] : 0.f;
#pragma unroll
    for (int v = 0; v < 8; ++v) {
      const int row = m0 + half * 8 + v;
      C[(size_t)row * N + col] = act_apply(acc[j][v] + bv, act);
    }
  }
}

// ------------------------------------------------- fused GATv2 logit kernel
// ee = e2@We (WMMA); m = leaky(xl[src]+xr[dst]+ee); logit[e,h] += dot(m, att[h])
// grid = (8 /*512/64*/, ceil(E2/128)), block = 256; dynamic LDS = 64*136*2.
__global__ __launch_bounds__(256)
void gat_logit_kernel(const u16* __restrict__ Ebf,   // [E2 x 128] bf16
                      const u16* __restrict__ WeT,   // [512 x 128] bf16
                      const float* __restrict__ xl,  // [M x 512]
                      const float* __restrict__ xr,  // [M x 512]
                      const float* __restrict__ att, // [512] (h*128+c)
                      const int* __restrict__ src2, const int* __restrict__ dst2,
                      float* __restrict__ logit,     // [E2 x 4]
                      int E2) {
  extern __shared__ u16 ldsB[];
  const int ldsStride = 128 + 8;
  const int n0 = blockIdx.x << 6;
  stage_B_panel(ldsB, ldsStride, WeT, n0, 128);
  __syncthreads();

  const int wave = threadIdx.x >> 5, lane = threadIdx.x & 31;
  const int half = lane >> 4, r = lane & 15;
  const int m0 = (blockIdx.y * 8 + wave) << 4;
  if (m0 >= E2) return;

  const v8f z = {0.f, 0.f, 0.f, 0.f, 0.f, 0.f, 0.f, 0.f};
  v8f acc[4] = {z, z, z, z};
  wmma_strip(acc, Ebf + (size_t)(m0 + r) * 128, ldsB, ldsStride, 128, half, r);

  int sidx[8], didx[8];
#pragma unroll
  for (int v = 0; v < 8; ++v) {
    const int row = m0 + half * 8 + v;
    sidx[v] = src2[row];
    didx[v] = dst2[row];
  }
#pragma unroll
  for (int j = 0; j < 4; ++j) {
    const int col = n0 + j * 16 + r;   // all 16 cols of this tile share one head
    const float av = att[col];
    const int head = col >> 7;
#pragma unroll
    for (int v = 0; v < 8; ++v) {
      float mv = acc[j][v] + xl[(size_t)sidx[v] * 512 + col] + xr[(size_t)didx[v] * 512 + col];
      mv = mv > 0.f ? mv : 0.01f * mv;                 // leaky_relu
      float c = mv * av;
      c += __shfl_xor(c, 1, 16);
      c += __shfl_xor(c, 2, 16);
      c += __shfl_xor(c, 4, 16);
      c += __shfl_xor(c, 8, 16);                       // reduce the 16 N-lanes
      if (r == 0) atomicAdd(&logit[(size_t)(m0 + half * 8 + v) * 4 + head], c);
    }
  }
}

// ------------------------------------------------- layernorm + elu (+bf16 copy)
__global__ __launch_bounds__(256)
void ln_elu_kernel(const float* __restrict__ x, const float* __restrict__ g,
                   const float* __restrict__ b, float* __restrict__ hn,
                   u16* __restrict__ hbf, long M) {
  const int wave = threadIdx.x >> 5, lane = threadIdx.x & 31;
  const long row = (long)blockIdx.x * 8 + wave;
  if (row >= M) return;
  const float* xr = x + row * 128;
  float v[4], s = 0.f, s2 = 0.f;
#pragma unroll
  for (int i = 0; i < 4; ++i) { v[i] = xr[lane + 32 * i]; s += v[i]; s2 += v[i] * v[i]; }
#pragma unroll
  for (int off = 16; off > 0; off >>= 1) {
    s  += __shfl_xor(s,  off, 32);
    s2 += __shfl_xor(s2, off, 32);
  }
  const float mu  = s * (1.f / 128.f);
  const float var = s2 * (1.f / 128.f) - mu * mu;
  const float inv = rsqrtf(var + 1e-5f);
#pragma unroll
  for (int i = 0; i < 4; ++i) {
    const int c = lane + 32 * i;
    float y = (v[i] - mu) * inv * g[c] + b[c];
    y = y > 0.f ? y : (__expf(y) - 1.f);              // elu
    hn[row * 128 + c] = y;
    hbf[row * 128 + c] = f2bf(y);
  }
}

// ------------------------------------------------- graph scatter kernels
__global__ void deg_kernel(const int* __restrict__ dst, float* __restrict__ deg, size_t E) {
  GS_LOOP(i, E) atomicAdd(&deg[dst[i]], 1.f);
}
__global__ void scatter_feat_kernel(const float* __restrict__ ea, const int* __restrict__ dst,
                                    float* __restrict__ acc, size_t total) {
  GS_LOOP(i, total) {
    size_t e = i >> 7; int c = (int)(i & 127);
    atomicAdd(&acc[(size_t)dst[e] * 128 + c], ea[i]);
  }
}
__global__ void mean_fin_kernel(float* __restrict__ acc, const float* __restrict__ deg, size_t total) {
  GS_LOOP(i, total) acc[i] /= fmaxf(deg[i >> 7], 1.f);
}
__global__ void build_idx2_kernel(const int* __restrict__ src, const int* __restrict__ dst,
                                  int* __restrict__ src2, int* __restrict__ dst2, int E, int M) {
  GS_LOOP(i, (size_t)(E + M)) {
    if (i < (size_t)E) { src2[i] = src[i]; dst2[i] = dst[i]; }
    else               { src2[i] = (int)i - E; dst2[i] = (int)i - E; }
  }
}
__global__ void segmax_kernel(const float* __restrict__ logit, const int* __restrict__ dst2,
                              float* __restrict__ maxb, size_t total) {
  GS_LOOP(i, total) {
    size_t e = i >> 2; int h = (int)(i & 3);
    atomicMaxF(&maxb[(size_t)dst2[e] * 4 + h], logit[i]);
  }
}
__global__ void expsum_kernel(float* __restrict__ logit, const int* __restrict__ dst2,
                              const float* __restrict__ maxb, float* __restrict__ sumb,
                              size_t total) {
  GS_LOOP(i, total) {
    size_t e = i >> 2; int h = (int)(i & 3);
    int d = dst2[e];
    float ex = __expf(logit[i] - maxb[(size_t)d * 4 + h]);
    logit[i] = ex;
    atomicAdd(&sumb[(size_t)d * 4 + h], ex);
  }
}
__global__ void aggregate_kernel(const float* __restrict__ ex, const float* __restrict__ sumb,
                                 const float* __restrict__ xl, const int* __restrict__ src2,
                                 const int* __restrict__ dst2, float* __restrict__ out,
                                 size_t total) {
  GS_LOOP(i, total) {
    size_t e = i >> 9; int c = (int)(i & 511); int h = c >> 7;
    int d = dst2[e], s = src2[e];
    float alpha = ex[e * 4 + h] / sumb[(size_t)d * 4 + h];
    atomicAdd(&out[(size_t)d * 512 + c], xl[(size_t)s * 512 + c] * alpha);
  }
}
__global__ void glu_kernel(const float* __restrict__ t, const float* __restrict__ resid,
                           float* __restrict__ out, size_t rows, int act) {
  GS_LOOP(i, rows * 128) {
    size_t m = i >> 7; int c = (int)(i & 127);
    float v = t[m * 256 + c], g = t[m * 256 + 128 + c];
    float X = v * act_apply(g, act);
    out[i] = (resid ? resid[i] : 0.f) + X;
  }
}
__global__ void pool_kernel(const float* __restrict__ h, const int* __restrict__ batch,
                            float* __restrict__ pooled, size_t total) {
  GS_LOOP(i, total) {
    size_t m = i >> 7; int c = (int)(i & 127);
    atomicAdd(&pooled[(size_t)batch[m] * 128 + c], h[i]);
  }
}
__global__ void outdot_kernel(const float* __restrict__ s, const float* __restrict__ W,
                              const float* __restrict__ b, float* __restrict__ out) {
  const int row = blockIdx.x, lane = threadIdx.x;
  float acc = 0.f;
  for (int c = lane; c < 128; c += 32) acc += s[row * 128 + c] * W[c];
#pragma unroll
  for (int off = 16; off > 0; off >>= 1) acc += __shfl_xor(acc, off, 32);
  if (lane == 0) out[row] = acc + b[0];
}

// ---------------------------------------------------------------- host side
static inline dim3 gs_grid(size_t total) {
  size_t nb = (total + 255) / 256;
  if (nb > 131072) nb = 131072;
  if (nb == 0) nb = 1;
  return dim3((unsigned)nb);
}
static inline size_t gemm_lds_bytes(int Kp) { return (size_t)64 * (Kp + 8) * 2; }

struct BlkP {
  const float *norm_g, *norm_b, *Wl, *bl, *Wr, *br, *We, *att, *gat_b, *gate_W, *gate_b;
};

static void run_block(float* xfeat, long M, const int* src, const int* dst, long E,
                      const float* eattr, const BlkP& p, char* ws, hipStream_t stream) {
  float* hn    = (float*)(ws + O_HN);
  float* xl    = (float*)(ws + O_XL);
  float* xr    = (float*)(ws + O_XR);
  float* gout  = (float*)(ws + O_GOUT);
  u16*   acta  = (u16*)(ws + O_ACTA);
  u16*   catb  = (u16*)(ws + O_CAT);
  float* meanb = (float*)(ws + O_CAT);   // aliased with catb: disjoint lifetime
  u16*   wt    = (u16*)(ws + O_WT);
  float* logit = (float*)(ws + O_LOGIT);
  float* maxb  = (float*)(ws + O_MAXB);
  float* sumb  = (float*)(ws + O_SUMB);
  float* deg   = (float*)(ws + O_DEG);
  int*   src2  = (int*)(ws + O_SRC2);
  int*   dst2  = (int*)(ws + O_DST2);
  const long E2 = E + M;
  const unsigned gy = (unsigned)((M + 127) / 128);

  // hn = elu(layernorm(x)); bf16 copy into acta
  ln_elu_kernel<<<dim3((unsigned)((M + 7) / 8)), 256, 0, stream>>>(
      xfeat, p.norm_g, p.norm_b, hn, acta, M);

  // xl = hn@Wl + bl ; xr = hn@Wr + br  (WMMA)
  transpose_cvt_kernel<<<gs_grid(512 * 128), 256, 0, stream>>>(wt, p.Wl, 128, 512, 128);
  wmma_gemm_kernel<<<dim3(8, gy), 256, gemm_lds_bytes(128), stream>>>(
      acta, wt, p.bl, xl, (int)M, 512, 128, ACT_NONE);
  transpose_cvt_kernel<<<gs_grid(512 * 128), 256, 0, stream>>>(wt, p.Wr, 128, 512, 128);
  wmma_gemm_kernel<<<dim3(8, gy), 256, gemm_lds_bytes(128), stream>>>(
      acta, wt, p.br, xr, (int)M, 512, 128, ACT_NONE);

  // mean edge feature per destination (self-loop fill_value='mean')
  fill_kernel<<<gs_grid(M), 256, 0, stream>>>(deg, 0.f, (size_t)M);
  fill_kernel<<<gs_grid((size_t)M * 128), 256, 0, stream>>>(meanb, 0.f, (size_t)M * 128);
  deg_kernel<<<gs_grid(E), 256, 0, stream>>>(dst, deg, (size_t)E);
  scatter_feat_kernel<<<gs_grid((size_t)E * 128), 256, 0, stream>>>(eattr, dst, meanb, (size_t)E * 128);
  mean_fin_kernel<<<gs_grid((size_t)M * 128), 256, 0, stream>>>(meanb, deg, (size_t)M * 128);

  // e2 = [eattr ; mean_e] in bf16; src2/dst2 with self loops appended
  cvt_rows_kernel<<<gs_grid((size_t)E * 128), 256, 0, stream>>>(
      acta, 128, 0, eattr, 128, nullptr, (size_t)E, 128, 128);
  cvt_rows_kernel<<<gs_grid((size_t)M * 128), 256, 0, stream>>>(
      acta + (size_t)E * 128, 128, 0, meanb, 128, nullptr, (size_t)M, 128, 128);
  build_idx2_kernel<<<gs_grid((size_t)E2), 256, 0, stream>>>(src, dst, src2, dst2, (int)E, (int)M);

  // fused: ee = e2@We (WMMA) + gather + leaky + att-dot -> logit[E2 x 4]
  transpose_cvt_kernel<<<gs_grid(512 * 128), 256, 0, stream>>>(wt, p.We, 128, 512, 128);
  fill_kernel<<<gs_grid((size_t)E2 * 4), 256, 0, stream>>>(logit, 0.f, (size_t)E2 * 4);
  gat_logit_kernel<<<dim3(8, (unsigned)((E2 + 127) / 128)), 256, gemm_lds_bytes(128), stream>>>(
      acta, wt, xl, xr, p.att, src2, dst2, logit, (int)E2);

  // segment softmax over dst2
  fill_kernel<<<gs_grid((size_t)M * 4), 256, 0, stream>>>(maxb, -3.0e38f, (size_t)M * 4);
  segmax_kernel<<<gs_grid((size_t)E2 * 4), 256, 0, stream>>>(logit, dst2, maxb, (size_t)E2 * 4);
  fill_kernel<<<gs_grid((size_t)M * 4), 256, 0, stream>>>(sumb, 0.f, (size_t)M * 4);
  expsum_kernel<<<gs_grid((size_t)E2 * 4), 256, 0, stream>>>(logit, dst2, maxb, sumb, (size_t)E2 * 4);

  // gout[dst] += xl[src] * alpha
  fill_kernel<<<gs_grid((size_t)M * 512), 256, 0, stream>>>(gout, 0.f, (size_t)M * 512);
  aggregate_kernel<<<gs_grid((size_t)E2 * 512), 256, 0, stream>>>(
      logit, sumb, xl, src2, dst2, gout, (size_t)E2 * 512);

  // GLU(cat(hn, gout + gat_b)) with elu, then residual with hn
  cvt_rows_kernel<<<gs_grid((size_t)M * 128), 256, 0, stream>>>(
      catb, 640, 0, hn, 128, nullptr, (size_t)M, 128, 128);
  cvt_rows_kernel<<<gs_grid((size_t)M * 512), 256, 0, stream>>>(
      catb, 640, 128, gout, 512, p.gat_b, (size_t)M, 512, 512);
  transpose_cvt_kernel<<<gs_grid((size_t)256 * 640), 256, 0, stream>>>(wt, p.gate_W, 640, 256, 640);
  float* t = xr;  // reuse (xr dead after gat_logit)
  wmma_gemm_kernel<<<dim3(4, gy), 256, gemm_lds_bytes(640), stream>>>(
      catb, wt, p.gate_b, t, (int)M, 256, 640, ACT_NONE);
  glu_kernel<<<gs_grid((size_t)M * 128), 256, 0, stream>>>(t, hn, xfeat, (size_t)M, ACT_ELU);
}

static inline BlkP slice_blk(void* const* d_in, int base, int layer) {
  BlkP p;
  p.norm_g = (const float*)d_in[base + 0] + (size_t)layer * 128;
  p.norm_b = (const float*)d_in[base + 1] + (size_t)layer * 128;
  p.Wl     = (const float*)d_in[base + 2] + (size_t)layer * 128 * 512;
  p.bl     = (const float*)d_in[base + 3] + (size_t)layer * 512;
  p.Wr     = (const float*)d_in[base + 4] + (size_t)layer * 128 * 512;
  p.br     = (const float*)d_in[base + 5] + (size_t)layer * 512;
  p.We     = (const float*)d_in[base + 6] + (size_t)layer * 128 * 512;
  p.att    = (const float*)d_in[base + 7] + (size_t)layer * 512;
  p.gat_b  = (const float*)d_in[base + 8] + (size_t)layer * 512;
  p.gate_W = (const float*)d_in[base + 9] + (size_t)layer * 640 * 256;
  p.gate_b = (const float*)d_in[base + 10] + (size_t)layer * 256;
  return p;
}

extern "C" void kernel_launch(void* const* d_in, const int* in_sizes, int n_in,
                              void* d_out, int out_size, void* d_ws, size_t ws_size,
                              hipStream_t stream) {
  (void)in_sizes; (void)n_in; (void)out_size; (void)ws_size;
  const float* x          = (const float*)d_in[0];
  const int*   eidx       = (const int*)d_in[1];
  const float* eattr_in   = (const float*)d_in[2];
  const int*   eidxL      = (const int*)d_in[3];
  const float* eattrL_in  = (const float*)d_in[4];
  const int*   batch      = (const int*)d_in[5];
  const int*   batchL     = (const int*)d_in[6];
  const float* gfeat      = (const float*)d_in[7];
  const float* pre_node_W = (const float*)d_in[8];
  const float* pre_node_b = (const float*)d_in[9];
  const float* pre_edge_W = (const float*)d_in[10];
  const float* pre_edge_b = (const float*)d_in[11];
  const float* pre_eline_W = (const float*)d_in[12];
  const float* pre_eline_b = (const float*)d_in[13];
  const float* glob_W      = (const float*)d_in[36];
  const float* pool_gate_W = (const float*)d_in[37];
  const float* pool_gate_b = (const float*)d_in[38];
  const float* post_W      = (const float*)d_in[39];
  const float* post_b      = (const float*)d_in[40];
  const float* out_W       = (const float*)d_in[41];
  const float* out_b       = (const float*)d_in[42];

  char* ws = (char*)d_ws;
  const int* src  = eidx;
  const int* dstn = eidx + cE;
  const int* srcL = eidxL;
  const int* dstL = eidxL + cEL;

  float* h      = (float*)(ws + O_H);
  float* xline  = (float*)(ws + O_XLINE);
  float* ea     = (float*)(ws + O_EA);
  float* eal    = (float*)(ws + O_EAL);
  u16*   acta   = (u16*)(ws + O_ACTA);
  u16*   wt     = (u16*)(ws + O_WT);

  // ---------------- pre-MLPs (silu) ----------------
  cvt_rows_kernel<<<gs_grid((size_t)cN * 96), 256, 0, stream>>>(
      acta, 96, 0, x, 92, nullptr, (size_t)cN, 92, 96);
  transpose_cvt_kernel<<<gs_grid((size_t)128 * 96), 256, 0, stream>>>(wt, pre_node_W, 92, 128, 96);
  wmma_gemm_kernel<<<dim3(2, (unsigned)((cN + 127) / 128)), 256, gemm_lds_bytes(96), stream>>>(
      acta, wt, pre_node_b, h, (int)cN, 128, 96, ACT_SILU);

  cvt_rows_kernel<<<gs_grid((size_t)cE * 64), 256, 0, stream>>>(
      acta, 64, 0, eattr_in, 50, nullptr, (size_t)cE, 50, 64);
  transpose_cvt_kernel<<<gs_grid((size_t)128 * 64), 256, 0, stream>>>(wt, pre_edge_W, 50, 128, 64);
  wmma_gemm_kernel<<<dim3(2, (unsigned)((cE + 127) / 128)), 256, gemm_lds_bytes(64), stream>>>(
      acta, wt, pre_edge_b, ea, (int)cE, 128, 64, ACT_SILU);

  cvt_rows_kernel<<<gs_grid((size_t)cEL * 64), 256, 0, stream>>>(
      acta, 64, 0, eattrL_in, 40, nullptr, (size_t)cEL, 40, 64);
  transpose_cvt_kernel<<<gs_grid((size_t)128 * 64), 256, 0, stream>>>(wt, pre_eline_W, 40, 128, 64);
  wmma_gemm_kernel<<<dim3(2, (unsigned)((cEL + 127) / 128)), 256, gemm_lds_bytes(64), stream>>>(
      acta, wt, pre_eline_b, eal, (int)cEL, 128, 64, ACT_SILU);

  copy_kernel<<<gs_grid((size_t)cE * 128), 256, 0, stream>>>(xline, ea, (size_t)cE * 128);

  // ---------------- GAT blocks ----------------
  for (int i = 0; i < 4; ++i)
    run_block(h, cN, src, dstn, cE, ea, slice_blk(d_in, 14, i), ws, stream);
  run_block(xline, cE, srcL, dstL, cEL, eal, slice_blk(d_in, 25, 0), ws, stream);

  // ---------------- pooling + head ----------------
  float* pooled  = (float*)(ws + O_POOL);
  float* pooledl = (float*)(ws + O_POOLL);
  float* gemb    = (float*)(ws + O_GEMB);
  float* t2      = (float*)(ws + O_T2);
  float* s0      = (float*)(ws + O_S0);
  float* s1      = (float*)(ws + O_S1);
  u16*   cat3    = (u16*)(ws + O_CAT3);

  fill_kernel<<<gs_grid((size_t)cB * 128), 256, 0, stream>>>(pooled, 0.f, (size_t)cB * 128);
  fill_kernel<<<gs_grid((size_t)cB * 128), 256, 0, stream>>>(pooledl, 0.f, (size_t)cB * 128);
  pool_kernel<<<gs_grid((size_t)cN * 128), 256, 0, stream>>>(h, batch, pooled, (size_t)cN * 128);
  pool_kernel<<<gs_grid((size_t)cE * 128), 256, 0, stream>>>(xline, batchL, pooledl, (size_t)cE * 128);

  cvt_rows_kernel<<<gs_grid((size_t)cB * 64), 256, 0, stream>>>(
      acta, 64, 0, gfeat, 64, nullptr, (size_t)cB, 64, 64);
  transpose_cvt_kernel<<<gs_grid((size_t)128 * 64), 256, 0, stream>>>(wt, glob_W, 64, 128, 64);
  wmma_gemm_kernel<<<dim3(2, 1), 256, gemm_lds_bytes(64), stream>>>(
      acta, wt, nullptr, gemb, (int)cB, 128, 64, ACT_NONE);

  cvt_rows_kernel<<<gs_grid((size_t)cB * 128), 256, 0, stream>>>(
      cat3, 384, 0, pooled, 128, nullptr, (size_t)cB, 128, 128);
  cvt_rows_kernel<<<gs_grid((size_t)cB * 128), 256, 0, stream>>>(
      cat3, 384, 128, pooledl, 128, nullptr, (size_t)cB, 128, 128);
  cvt_rows_kernel<<<gs_grid((size_t)cB * 128), 256, 0, stream>>>(
      cat3, 384, 256, gemb, 128, nullptr, (size_t)cB, 128, 128);
  transpose_cvt_kernel<<<gs_grid((size_t)256 * 384), 256, 0, stream>>>(wt, pool_gate_W, 384, 256, 384);
  wmma_gemm_kernel<<<dim3(4, 1), 256, gemm_lds_bytes(384), stream>>>(
      cat3, wt, pool_gate_b, t2, (int)cB, 256, 384, ACT_NONE);
  glu_kernel<<<gs_grid((size_t)cB * 128), 256, 0, stream>>>(t2, nullptr, s0, (size_t)cB, ACT_SILU);

  float* scur = s0; float* snext = s1;
  for (int i = 0; i < 2; ++i) {
    cvt_rows_kernel<<<gs_grid((size_t)cB * 128), 256, 0, stream>>>(
        acta, 128, 0, scur, 128, nullptr, (size_t)cB, 128, 128);
    transpose_cvt_kernel<<<gs_grid((size_t)128 * 128), 256, 0, stream>>>(
        wt, post_W + (size_t)i * 128 * 128, 128, 128, 128);
    wmma_gemm_kernel<<<dim3(2, 1), 256, gemm_lds_bytes(128), stream>>>(
        acta, wt, post_b + (size_t)i * 128, snext, (int)cB, 128, 128, ACT_SILU);
    float* tmp = scur; scur = snext; snext = tmp;
  }
  outdot_kernel<<<dim3((unsigned)cB), 32, 0, stream>>>(scur, out_W, out_b, (float*)d_out);
}
